// KGFlexModel_58136677319049
// MI455X (gfx1250) — matrix operand order repaired and checked by reading, not compiled, not run on hardware
//
#include <hip/hip_runtime.h>

// ---------------------------------------------------------------------------
// KGFlex fused kernel for MI455X (gfx1250, wave32, WMMA)
//   x[b] = sum_f K[user[b],f] * (C[item[b],f]-1) * (H[user[b]].G[f] + F_B[f])
//          + I_B[item[b]]
// GEMM part (H @ G^T) via v_wmma_f32_16x16x32_bf16.  K/C row gathers stream
// fp32 from HBM (~512 MB -> ~22us floor at 23.3 TB/s) and bound the kernel.
// G is pre-converted to bf16 in d_ws (1 MB) so the inner loop loads WMMA B
// operands directly with b128s.  16 waves/block + global_prefetch of the K/C
// stream keep enough bytes in flight to feed 23.3 TB/s.
// ---------------------------------------------------------------------------

#define F_DIM 8192
#define D_DIM 64
#define WAVES_PER_BLOCK 16
#define BLOCK_THREADS (WAVES_PER_BLOCK * 32)
#define FCHUNK (F_DIM / WAVES_PER_BLOCK)   // 512 features per wave
#define PF_DIST 256                         // prefetch bytes ahead (4 steps)

typedef __attribute__((ext_vector_type(16))) __bf16 v16bf;
typedef __attribute__((ext_vector_type(8)))  float  v8f;

union BF16x16 { v16bf v; unsigned int u[8]; uint4 q[2]; };

// two fp32 -> dword of two bf16, round-to-nearest-even (prep / A build)
__device__ inline unsigned int pack2bf_rne(float lo, float hi) {
  unsigned int a = __builtin_bit_cast(unsigned int, lo);
  unsigned int b = __builtin_bit_cast(unsigned int, hi);
  a = (a + 0x7FFFu + ((a >> 16) & 1u)) >> 16;
  b = (b + 0x7FFFu + ((b >> 16) & 1u)) >> 16;
  return (b << 16) | (a & 0xFFFFu);
}

// two fp32 -> dword of two bf16, truncation: single v_perm_b32 (fallback)
__device__ inline unsigned int pack2bf_trunc(float lo, float hi) {
  return __builtin_amdgcn_perm(__builtin_bit_cast(unsigned int, hi),
                               __builtin_bit_cast(unsigned int, lo),
                               0x07060302u);
}

// butterfly add across the 16-lane halves via ds_swizzle (xor masks 1,2,4,8)
__device__ inline float half_wave_reduce(float s) {
  int x = __builtin_bit_cast(int, s);
#define SWZ(mask) \
  s += __builtin_bit_cast(float, __builtin_amdgcn_ds_swizzle(x, ((mask) << 10) | 0x1F)); \
  x = __builtin_bit_cast(int, s);
  SWZ(1) SWZ(2) SWZ(4) SWZ(8)
#undef SWZ
  return s;
}

// ---- prep: G (F x 64, fp32) -> Gbf (F x 64, bf16) in workspace ------------
__global__ __launch_bounds__(256)
void g_to_bf16_kernel(const float* __restrict__ G, unsigned int* __restrict__ Gbf) {
  const int i = blockIdx.x * blockDim.x + threadIdx.x;   // one dword = 2 elems
  if (i < F_DIM * D_DIM / 2)
    Gbf[i] = pack2bf_rne(G[2 * i], G[2 * i + 1]);
}

// ---- main fused kernel ----------------------------------------------------
template <bool USE_WS>
__global__ __launch_bounds__(BLOCK_THREADS)
void kgflex_kernel(const int* __restrict__ user, const int* __restrict__ item,
                   const float* __restrict__ F_B, const float* __restrict__ I_B,
                   const float* __restrict__ H,   const float* __restrict__ G,
                   const float* __restrict__ K,   const float* __restrict__ C,
                   const uint4* __restrict__ Gbf, float* __restrict__ out) {
  __shared__ float partial[WAVES_PER_BLOCK * 16];

  const int tid  = threadIdx.x;
  const int wave = tid >> 5;
  const int lane = tid & 31;
  const int lam  = lane & 15;   // column / A-row selector
  const int hi   = lane >> 4;   // half-wave selector
  const int b0   = blockIdx.x * 16;   // 16-row b tile for the whole block
  const int f0   = wave * FCHUNK;     // this wave's feature range

  // A operands: gathered H rows, bf16, contiguous k order, same as B.
  // lane (lam,hi) supplies row M=lam; lo half K {0..15}+k0,
  // hi half K {16..31}+k0; two operands cover D=64.
  const float* hrow = H + (long)user[b0 + lam] * D_DIM;
  BF16x16 A0, A1;
#pragma unroll
  for (int r = 0; r < 4; ++r) {
    const int d = 16 * hi + 2 * r;
    A0.u[r]     = pack2bf_rne(hrow[d],      hrow[d + 1]);
    A0.u[r + 4] = pack2bf_rne(hrow[d + 8],  hrow[d + 9]);
    A1.u[r]     = pack2bf_rne(hrow[d + 32], hrow[d + 33]);
    A1.u[r + 4] = pack2bf_rne(hrow[d + 40], hrow[d + 41]);
  }

  // z C-layout: VGPR v, lane -> row m = v + 8*hi, col f = fbase + lam
  int offK[8], offC[8];
#pragma unroll
  for (int v = 0; v < 8; ++v) {
    const int b = b0 + v + 8 * hi;
    offK[v] = user[b] * F_DIM;
    offC[v] = item[b] * F_DIM;
  }

  float acc[8];
#pragma unroll
  for (int v = 0; v < 8; ++v) acc[v] = 0.0f;

  for (int fb = f0; fb < f0 + FCHUNK; fb += 16) {
    const int f = fb + lam;

    // prefetch the K/C stream ahead (speculative; OOB silently dropped)
#pragma unroll
    for (int v = 0; v < 8; ++v) {
      __builtin_prefetch((const char*)(K + offK[v] + f) + PF_DIST, 0, 3);
      __builtin_prefetch((const char*)(C + offC[v] + f) + PF_DIST, 0, 3);
    }

    // B operands: G[f, d] -> B matrix (K=d, N=f). lane (lam,hi) = column
    // N=lam, K rows d = 16*hi + {0..15} (+32 for the second operand).
    BF16x16 B0, B1;
    if constexpr (USE_WS) {
      const uint4* gb = Gbf + (size_t)f * (D_DIM / 8) + 2 * hi; // 8 bf16/uint4
      B0.q[0] = gb[0];
      B0.q[1] = gb[1];
      B1.q[0] = gb[4];
      B1.q[1] = gb[5];
    } else {
      const float* grow = G + (long)f * D_DIM + 16 * hi;
#pragma unroll
      for (int r = 0; r < 8; ++r) {
        B0.u[r] = pack2bf_trunc(grow[2 * r],      grow[2 * r + 1]);
        B1.u[r] = pack2bf_trunc(grow[2 * r + 32], grow[2 * r + 33]);
      }
    }

    // z = h_tile @ G_chunk^T  (16x16 f32 tile, K-dim D=64 via 2 WMMAs)
    v8f z = {};
    z = __builtin_amdgcn_wmma_f32_16x16x32_bf16(false, A0.v, false, B0.v,
                                                (short)0, z, false, false);
    z = __builtin_amdgcn_wmma_f32_16x16x32_bf16(false, A1.v, false, B1.v,
                                                (short)0, z, false, false);

    const float fbv = F_B[f];
#pragma unroll
    for (int v = 0; v < 8; ++v) {
      const float kf = K[offK[v] + f];   // HBM stream, 64B-coalesced per row
      const float cf = C[offC[v] + f];
      const float kc = fmaf(kf, cf, -kf);        // K * (C - 1)
      acc[v] = fmaf(kc, z[v] + fbv, acc[v]);
    }
  }

  // reduce over the 16 feature-columns held across each half-wave
#pragma unroll
  for (int v = 0; v < 8; ++v) {
    const float s = half_wave_reduce(acc[v]);
    if (lam == 0) partial[wave * 16 + v + 8 * hi] = s;
  }
  __syncthreads();

  // combine the waves' partials, add item bias, write out
  if (tid < 16) {
    float s = 0.0f;
#pragma unroll
    for (int w = 0; w < WAVES_PER_BLOCK; ++w) s += partial[w * 16 + tid];
    out[b0 + tid] = s + I_B[item[b0 + tid]];
  }
}

// ---------------------------------------------------------------------------
extern "C" void kernel_launch(void* const* d_in, const int* in_sizes, int n_in,
                              void* d_out, int out_size, void* d_ws, size_t ws_size,
                              hipStream_t stream) {
  const int*   user = (const int*)  d_in[0];
  const int*   item = (const int*)  d_in[1];
  const float* F_B  = (const float*)d_in[2];
  const float* I_B  = (const float*)d_in[3];
  const float* H    = (const float*)d_in[4];
  const float* G    = (const float*)d_in[5];
  const float* K    = (const float*)d_in[6];
  const float* C    = (const float*)d_in[7];
  float* out = (float*)d_out;

  const int nblocks = (out_size + 15) / 16;               // 512 b-tiles
  const size_t needG = (size_t)F_DIM * D_DIM * 2;         // 1 MB bf16 G

  if (ws_size >= needG) {
    // per-launch fp32 -> bf16 conversion of G into workspace (tiny)
    const int ndw = F_DIM * D_DIM / 2;
    g_to_bf16_kernel<<<(ndw + 255) / 256, 256, 0, stream>>>(
        G, (unsigned int*)d_ws);
    kgflex_kernel<true><<<nblocks, BLOCK_THREADS, 0, stream>>>(
        user, item, F_B, I_B, H, G, K, C, (const uint4*)d_ws, out);
  } else {
    kgflex_kernel<false><<<nblocks, BLOCK_THREADS, 0, stream>>>(
        user, item, F_B, I_B, H, G, K, C, nullptr, out);
  }
}
